// Attention_35364760715947
// MI455X (gfx1250) — matrix operand order repaired and checked by reading, not compiled, not run on hardware
//
#include <hip/hip_runtime.h>
#include <hip/hip_bf16.h>
#include <math.h>

// Problem dims (compile-time, from reference)
#define BB    64
#define TT    2048
#define EMB   512
#define DEC   1024
#define ADIM  128
#define NFILT 32
#define KSIZE 31

#define WMP_ELEMS ((EMB / 32) * 8 * 32 * 16)   // 65536 bf16 (128 KB)
#define WLP_ELEMS (8 * 32 * 16)                // 4096 bf16 (8 KB)
#define SMEM_BYTES ((WMP_ELEMS + WLP_ELEMS) * 2)  // 139264 B dynamic LDS

typedef __attribute__((ext_vector_type(16))) __bf16 v16bf;
typedef __attribute__((ext_vector_type(8)))  float  v8f;

static __device__ __forceinline__ v8f wmma_bf16(v16bf a, v16bf b, v8f c) {
  // D = A(16x32 bf16) x B(32x16 bf16) + C(16x16 f32)
  return __builtin_amdgcn_wmma_f32_16x16x32_bf16(
      /*neg_a=*/false, a, /*neg_b=*/false, b,
      /*c_mod=*/(short)0, c, /*reuse_a=*/false, /*reuse_b=*/false);
}

// Fill a 16xK A-fragment (lane m = l&15, half h = l>>4) from a row pointer:
// element e -> k = e + 16*h, i.e. one contiguous 16-float run.
static __device__ __forceinline__ v16bf load_afrag_row(const float* __restrict__ row,
                                                       int hh) {
  float4 f0 = *(const float4*)(row + 16 * hh);
  float4 f1 = *(const float4*)(row + 16 * hh + 4);
  float4 f2 = *(const float4*)(row + 16 * hh + 8);
  float4 f3 = *(const float4*)(row + 16 * hh + 12);
  v16bf a;
  a[0]  = (__bf16)f0.x; a[1]  = (__bf16)f0.y; a[2]  = (__bf16)f0.z; a[3]  = (__bf16)f0.w;
  a[4]  = (__bf16)f1.x; a[5]  = (__bf16)f1.y; a[6]  = (__bf16)f1.z; a[7]  = (__bf16)f1.w;
  a[8]  = (__bf16)f2.x; a[9]  = (__bf16)f2.y; a[10] = (__bf16)f2.z; a[11] = (__bf16)f2.w;
  a[12] = (__bf16)f3.x; a[13] = (__bf16)f3.y; a[14] = (__bf16)f3.z; a[15] = (__bf16)f3.w;
  return a;
}

// ---------------------------------------------------------------------------
// Kernel 1: qsum[b][a] = query[b]·Wq[:,a] + bq[a] + bm[a] + bl[a]
// ---------------------------------------------------------------------------
__global__ void qsum_kernel(const float* __restrict__ query,
                            const float* __restrict__ Wq,
                            const float* __restrict__ bq,
                            const float* __restrict__ bm,
                            const float* __restrict__ bl,
                            float* __restrict__ qsum) {
  int idx = blockIdx.x * blockDim.x + threadIdx.x;
  if (idx >= BB * ADIM) return;
  int b = idx / ADIM, a = idx % ADIM;
  float s = bq[a] + bm[a] + bl[a];
  const float* qr = query + (size_t)b * DEC;
  #pragma unroll 8
  for (int d = 0; d < DEC; ++d) s += qr[d] * Wq[(size_t)d * ADIM + a];
  qsum[idx] = s;
}

// ---------------------------------------------------------------------------
// Kernel 2: conv_out[b][t][f] = SAME conv1d(awc[b,:,0], conv_k[:,0,f]) + cb[f]
// ---------------------------------------------------------------------------
__global__ void conv_kernel(const float* __restrict__ awc,
                            const float* __restrict__ ck,
                            const float* __restrict__ cb,
                            float* __restrict__ conv_out) {
  int idx = blockIdx.x * blockDim.x + threadIdx.x;
  if (idx >= BB * TT * NFILT) return;
  int f = idx % NFILT;
  int t = (idx / NFILT) % TT;
  int b = idx / (NFILT * TT);
  const float* aw = awc + (size_t)b * TT;
  float s = cb[f];
  #pragma unroll
  for (int j = 0; j < KSIZE; ++j) {
    int tt = t + j - (KSIZE / 2);
    if (tt >= 0 && tt < TT) s += aw[tt] * ck[j * NFILT + f];
  }
  conv_out[idx] = s;
}

// ---------------------------------------------------------------------------
// Kernel 2b: pack Wm (512x128 f32) into bf16 B-fragment layout.
// wm_p[((kt*8 + n)*32 + lane)*16 + e] = bf16( Wm[kt*32 + e + 16*(lane>>4)][16n + (lane&15)] )
// One fragment load in the GEMM = one contiguous 32-byte v16bf per lane.
// ---------------------------------------------------------------------------
__global__ void pack_wm_kernel(const float* __restrict__ Wm,
                               __bf16* __restrict__ wm_p) {
  int idx = blockIdx.x * blockDim.x + threadIdx.x;   // 65536
  if (idx >= WMP_ELEMS) return;
  int e    = idx & 15;
  int lane = (idx >> 4) & 31;
  int n    = (idx >> 9) & 7;
  int kt   = idx >> 12;
  int k    = kt * 32 + e + 16 * (lane >> 4);
  wm_p[idx] = (__bf16)Wm[(size_t)k * ADIM + 16 * n + (lane & 15)];
}

// Kernel 2c: pack Wl (32x128 f32) likewise (single K-step).
__global__ void pack_wl_kernel(const float* __restrict__ Wl,
                               __bf16* __restrict__ wl_p) {
  int idx = blockIdx.x * blockDim.x + threadIdx.x;   // 4096
  if (idx >= WLP_ELEMS) return;
  int e    = idx & 15;
  int lane = (idx >> 4) & 31;
  int n    = idx >> 9;
  int k    = e + 16 * (lane >> 4);
  wl_p[idx] = (__bf16)Wl[k * ADIM + 16 * n + (lane & 15)];
}

// ---------------------------------------------------------------------------
// Kernel 3: energies via WMMA, with Wm+Wl fragments staged in LDS per block.
// Each wave: one 16-row tile of T, all 8 column tiles of ADIM.
//   acc[n] initialized from qsum (q + biases, broadcast over rows)
//   acc[n] += conv_out(16x32) @ Wl[:,16n:16n+16]        (1 WMMA, B from LDS)
//   acc[n] += memory(16x512)  @ Wm[:,16n:16n+16]        (16 WMMAs, B from LDS)
//   energy  = sum_a tanh(acc)*V[a]  (+bV, +mask*-1e9)   (shfl reduce)
// Block = 128 threads = 4 waves => 64 rows of T per block.
// Dynamic LDS = 139264 B (packed Wm 128KB + Wl 8KB); gfx1250 WGP has 320KB.
// ---------------------------------------------------------------------------
__global__ void __launch_bounds__(128, 1)
energies_kernel(const float* __restrict__ mem,
                const float* __restrict__ conv_out,
                const __bf16* __restrict__ wm_p,  // wl_p is contiguous after wm_p
                const float* __restrict__ Vvec,
                const float* __restrict__ bV,
                const float* __restrict__ mask,
                const float* __restrict__ qsum,
                float* __restrict__ ener) {
  extern __shared__ __bf16 smem[];   // [WMP_ELEMS + WLP_ELEMS]

  const int b      = blockIdx.x / (TT / 64);
  const int tchunk = blockIdx.x % (TT / 64);
  const int wave   = threadIdx.x >> 5;
  const int lane   = threadIdx.x & 31;
  const int hh     = lane >> 4;   // wave half (0: lanes 0-15, 1: lanes 16-31)
  const int n15    = lane & 15;   // position within half
  const int t0     = tchunk * 64 + wave * 16;

  // --- cooperative stage of packed Wm|Wl (contiguous 139264 B) into LDS ----
  {
    const uint4* src = (const uint4*)wm_p;
    uint4*       dst = (uint4*)smem;
    const int nvec = (WMP_ELEMS + WLP_ELEMS) * 2 / 16;  // 8704 x 16B
    for (int i = threadIdx.x; i < nvec; i += 128) dst[i] = src[i];
  }
  __syncthreads();
  const v16bf* wmf = (const v16bf*)smem;                 // fragment views (LDS)
  const v16bf* wlf = (const v16bf*)(smem + WMP_ELEMS);

  // --- init accumulators with q + biases (same for every row of the tile) ---
  v8f acc[8];
  #pragma unroll
  for (int n = 0; n < 8; ++n) {
    float qv = qsum[b * ADIM + 16 * n + n15];
    #pragma unroll
    for (int r = 0; r < 8; ++r) acc[n][r] = qv;
  }

  // --- loc projection: conv_out tile (16x32) @ Wl (32x128), single K-step ---
  {
    v16bf afrag = load_afrag_row(conv_out + ((size_t)(b * TT + t0 + n15)) * NFILT, hh);
    #pragma unroll
    for (int n = 0; n < 8; ++n) {
      v16bf bfrag = wlf[n * 32 + lane];
      acc[n] = wmma_bf16(afrag, bfrag, acc[n]);
    }
  }

  // --- main GEMM: memory tile (16x512) @ Wm (512x128), 16 K-steps of 32 ----
  const float* arow_base = mem + ((size_t)(b * TT + t0 + n15)) * EMB;
  for (int kt = 0; kt < EMB / 32; ++kt) {
    const float* arow = arow_base + kt * 32;
    __builtin_prefetch(arow + 32, 0, 0);  // global_prefetch_b8 for next K-tile
    v16bf afrag = load_afrag_row(arow, hh);
    #pragma unroll
    for (int n = 0; n < 8; ++n) {
      v16bf bfrag = wmf[(kt * 8 + n) * 32 + lane];
      acc[n] = wmma_bf16(afrag, bfrag, acc[n]);
    }
  }

  // --- epilogue: tanh, dot with V, reduce over the 16 lanes of each half ---
  float vv[8];
  #pragma unroll
  for (int n = 0; n < 8; ++n) vv[n] = Vvec[16 * n + n15];

  float part[8];
  #pragma unroll
  for (int r = 0; r < 8; ++r) {
    float s = 0.f;
    #pragma unroll
    for (int n = 0; n < 8; ++n) s += tanhf(acc[n][r]) * vv[n];
    part[r] = s;
  }
  #pragma unroll
  for (int off = 1; off < 16; off <<= 1) {
    #pragma unroll
    for (int r = 0; r < 8; ++r) part[r] += __shfl_xor(part[r], off, 32);
  }

  if (n15 == 0) {
    float b0 = bV[0];
    #pragma unroll
    for (int r = 0; r < 8; ++r) {
      int t = t0 + r + 8 * hh;  // rows 0-7 live in half 0, rows 8-15 in half 1
      ener[b * TT + t] = part[r] + b0 + mask[b * TT + t] * -1e9f;
    }
  }
}

// ---------------------------------------------------------------------------
// Kernel 4: softmax over T per batch row. weights -> d_out (weights section)
// ---------------------------------------------------------------------------
__global__ void softmax_kernel(const float* __restrict__ ener,
                               float* __restrict__ weights) {
  __shared__ float red[256];
  const int b   = blockIdx.x;
  const int tid = threadIdx.x;
  const float* e = ener + (size_t)b * TT;

  float m = -INFINITY;
  for (int t = tid; t < TT; t += 256) m = fmaxf(m, e[t]);
  red[tid] = m; __syncthreads();
  for (int s = 128; s > 0; s >>= 1) {
    if (tid < s) red[tid] = fmaxf(red[tid], red[tid + s]);
    __syncthreads();
  }
  m = red[0]; __syncthreads();

  float sum = 0.f;
  for (int t = tid; t < TT; t += 256) sum += __expf(e[t] - m);
  red[tid] = sum; __syncthreads();
  for (int s = 128; s > 0; s >>= 1) {
    if (tid < s) red[tid] += red[tid + s];
    __syncthreads();
  }
  float inv = 1.f / red[0];

  for (int t = tid; t < TT; t += 256)
    weights[(size_t)b * TT + t] = __expf(e[t] - m) * inv;
}

// ---------------------------------------------------------------------------
// Kernel 5: context[b][e] = sum_t w[b][t] * memory[b][t][e]
// ---------------------------------------------------------------------------
__global__ void context_kernel(const float* __restrict__ weights,
                               const float* __restrict__ mem,
                               float* __restrict__ ctx) {
  const int b = blockIdx.x / (EMB / 256);
  const int e = (blockIdx.x % (EMB / 256)) * 256 + threadIdx.x;
  const float* w  = weights + (size_t)b * TT;
  const float* mb = mem + (size_t)b * TT * EMB + e;
  float acc = 0.f;
  #pragma unroll 4
  for (int t = 0; t < TT; ++t) acc += w[t] * mb[(size_t)t * EMB];
  ctx[b * EMB + e] = acc;
}

// ---------------------------------------------------------------------------
extern "C" void kernel_launch(void* const* d_in, const int* in_sizes, int n_in,
                              void* d_out, int out_size, void* d_ws, size_t ws_size,
                              hipStream_t stream) {
  const float* query = (const float*)d_in[0];
  const float* mem   = (const float*)d_in[1];
  const float* awc   = (const float*)d_in[2];
  const float* mask  = (const float*)d_in[3];
  const float* Wq    = (const float*)d_in[4];
  const float* bq    = (const float*)d_in[5];
  const float* Wm    = (const float*)d_in[6];
  const float* bm    = (const float*)d_in[7];
  const float* ck    = (const float*)d_in[8];
  const float* cb    = (const float*)d_in[9];
  const float* Wl    = (const float*)d_in[10];
  const float* bl    = (const float*)d_in[11];
  const float* Vv    = (const float*)d_in[12];
  const float* bV    = (const float*)d_in[13];

  float* out_ctx = (float*)d_out;                   // [B, EMB]
  float* out_w   = (float*)d_out + BB * EMB;        // [B, T]

  // Workspace layout: ener[B*T] f32 | conv_out[B*T*NFILT] f32 | qsum[B*ADIM] f32
  //                 | wm_p[65536] bf16 | wl_p[4096] bf16  (wm_p/wl_p contiguous)
  float*  ws       = (float*)d_ws;
  float*  ener     = ws;
  float*  conv_out = ws + (size_t)BB * TT;
  float*  qsum     = conv_out + (size_t)BB * TT * NFILT;
  __bf16* wm_p     = (__bf16*)(qsum + (size_t)BB * ADIM);
  __bf16* wl_p     = wm_p + (size_t)WMP_ELEMS;

  qsum_kernel<<<(BB * ADIM + 255) / 256, 256, 0, stream>>>(query, Wq, bq, bm, bl, qsum);
  conv_kernel<<<(BB * TT * NFILT + 255) / 256, 256, 0, stream>>>(awc, ck, cb, conv_out);
  pack_wm_kernel<<<(WMP_ELEMS + 255) / 256, 256, 0, stream>>>(Wm, wm_p);
  pack_wl_kernel<<<(WLP_ELEMS + 255) / 256, 256, 0, stream>>>(Wl, wl_p);
  energies_kernel<<<BB * (TT / 64), 128, SMEM_BYTES, stream>>>(
      mem, conv_out, wm_p, Vv, bV, mask, qsum, ener);
  softmax_kernel<<<BB, 256, 0, stream>>>(ener, out_w);
  context_kernel<<<BB * (EMB / 256), 256, 0, stream>>>(out_w, mem, out_ctx);
}
